// ChemicalSpecialist2D_24378234372361
// MI455X (gfx1250) — compile-verified, matrix operand order checked by reading, not run on hardware
//
#include <hip/hip_runtime.h>

#define N_NODES 100000
#define N_EDGES 300000

typedef __attribute__((ext_vector_type(16))) _Float16 v16h;
typedef __attribute__((ext_vector_type(8)))  float    v8f;

__device__ __forceinline__ float frelu(float x) { return x > 0.f ? x : 0.f; }

__device__ __forceinline__ v8f wmma16(v16h a, v16h b, v8f c) {
  // v_wmma_f32_16x16x32_f16: D = A(16x32 f16) x B(32x16 f16) + C(16x16 f32)
  return __builtin_amdgcn_wmma_f32_16x16x32_f16(false, a, false, b, (short)0, c, false, false);
}

// Contiguous 32-byte fragment load -> 2x b128.
__device__ __forceinline__ v16h ldfrag(const _Float16* p) { return *(const v16h*)p; }

// Position of logical column/k `k` inside a fragment-ordered 32-wide block
// (A-matrix per-row layout): slot = (k&7) | ((k>>4)&1)<<3 ; half = (k>>3)&1 selects +16.
__device__ __forceinline__ int apos(int k) {
  return ((k >> 5) << 5) + (k & 7) + (((k >> 4) & 1) << 3) + (((k >> 3) & 1) << 4);
}

// ---------------- weight swizzle prep: KxN f32 row-major -> f16 panel-fragment order --------
// Per k-panel (32 rows): [n-tile of 16][lane (n&15 | kbit3<<4)][slot (k&7 | kbit4<<3)]
__global__ void k_prep_b(const float* __restrict__ src, _Float16* __restrict__ dst,
                         int K, int Kpad, int N) {
  int idx = blockIdx.x * 256 + threadIdx.x;
  if (idx >= Kpad * N) return;
  int k = idx / N, c = idx - k * N;
  float v = (k < K) ? src[k * N + c] : 0.f;
  int lane = (c & 15) + (((k >> 3) & 1) << 4);
  int slot = (k & 7) + (((k >> 4) & 1) << 3);
  dst[(size_t)(k >> 5) * 32 * N + (c >> 4) * 512 + lane * 16 + slot] = (_Float16)v;
}

// bond embedding table 5x64 -> A-fragment (per-row) order
__global__ void k_prep_be(const float* __restrict__ bond_tab, _Float16* __restrict__ be_sw) {
  int idx = blockIdx.x * 256 + threadIdx.x;
  if (idx >= 5 * 64) return;
  int bt = idx >> 6, k = idx & 63;
  be_sw[bt * 64 + apos(k)] = (_Float16)bond_tab[idx];
}

// ---------------- node init: atom emb gather, valence MLP, argmax, h0 ----------------
__global__ void k_node_init(const float* __restrict__ x, const float* __restrict__ atom_emb,
                            const float* __restrict__ vp_w1, const float* __restrict__ vp_b1,
                            const float* __restrict__ vp_w2, const float* __restrict__ vp_b2,
                            float* __restrict__ h, _Float16* __restrict__ h16,
                            int* __restrict__ pv, float* __restrict__ vlog_out) {
  int n = blockIdx.x * 256 + threadIdx.x;
  if (n >= N_NODES) return;
  int at = (int)x[n * 16];
  at = at < 0 ? 0 : (at > 10 ? 10 : at);
  float ae[64];
#pragma unroll
  for (int i = 0; i < 64; ++i) ae[i] = atom_emb[at * 64 + i];
  float hid[32];
  for (int j = 0; j < 32; ++j) {
    float s = vp_b1[j];
#pragma unroll 8
    for (int i = 0; i < 64; ++i) s += ae[i] * vp_w1[i * 32 + j];
    hid[j] = frelu(s);
  }
  float lg[8];
#pragma unroll
  for (int o = 0; o < 8; ++o) {
    float s = vp_b2[o];
#pragma unroll 8
    for (int j = 0; j < 32; ++j) s += hid[j] * vp_w2[j * 8 + o];
    lg[o] = s;
    vlog_out[n * 8 + o] = s;
  }
  int best = 0;
#pragma unroll
  for (int o = 1; o < 8; ++o) if (lg[o] > lg[best]) best = o;
  pv[n] = best + 1;
  size_t rb = (size_t)n * 256;
#pragma unroll
  for (int c = 0; c < 64; ++c) {
    float v = ae[c];
    h[rb + c] = v;
    h16[rb + apos(c)] = (_Float16)v;
  }
  for (int c = 64; c < 256; ++c) {
    float v = (c == 64 + best) ? 1.f : 0.f;
    h[rb + c] = v;
    h16[rb + apos(c)] = (_Float16)v;
  }
}

// ---------------- message GEMM + scatter: agg[row] += relu([h[col]|be] @ Wmsg + b) ----------
// No LDS: A fragments gathered directly from swizzled h16 / be tables; B direct from
// swizzled weights (L2-resident).
__global__ void __launch_bounds__(256) k_msg(const _Float16* __restrict__ h16,
                                             const int* __restrict__ edge_index,
                                             const float* __restrict__ edge_attr,
                                             const _Float16* __restrict__ be_sw,
                                             const _Float16* __restrict__ wsw,
                                             const float* __restrict__ bmsg,
                                             float* __restrict__ agg) {
  int tid = threadIdx.x, tile = blockIdx.x;
  int wv = tid >> 5, lane = tid & 31;
  int mlane = lane & 15, hi16 = (lane >> 4) << 4;
  int em = tile * 16 + mlane;
  int cnode = edge_index[N_EDGES + em];
  int bt = (int)edge_attr[em * 4];
  bt = bt < 0 ? 0 : (bt > 4 ? 4 : bt);
  const _Float16* arow  = h16 + (size_t)cnode * 256 + hi16;
  const _Float16* berow = be_sw + bt * 64 + hi16;
  const _Float16* bbase = wsw + (size_t)(wv * 2) * 512 + lane * 16;
  v8f acc0 = {}, acc1 = {};
#pragma unroll
  for (int ks = 0; ks < 10; ++ks) {
    v16h a = (ks < 8) ? ldfrag(arow + ks * 32) : ldfrag(berow + (ks - 8) * 32);
    const _Float16* bp = bbase + (size_t)ks * (32 * 256);
    acc0 = wmma16(a, ldfrag(bp), acc0);
    acc1 = wmma16(a, ldfrag(bp + 512), acc1);
  }
  int n0 = wv * 32 + mlane;
  float b0 = bmsg[n0], b1 = bmsg[n0 + 16];
  int moff = (lane >> 4) << 3;
#pragma unroll
  for (int v = 0; v < 8; ++v) {
    int r = edge_index[tile * 16 + v + moff];
    atomicAdd(&agg[(size_t)r * 256 + n0], frelu(acc0[v] + b0));
    atomicAdd(&agg[(size_t)r * 256 + n0 + 16], frelu(acc1[v] + b1));
  }
}

// ---------------- self GEMM: h = relu(h @ Wself + b + agg), in-place on h16 ----------------
__global__ void __launch_bounds__(256) k_self(const float* __restrict__ agg,
                                              const _Float16* __restrict__ wsw,
                                              const float* __restrict__ bself,
                                              float* __restrict__ h,
                                              _Float16* __restrict__ h16) {
  int tid = threadIdx.x, base = blockIdx.x * 16;
  int wv = tid >> 5, lane = tid & 31;
  int mlane = lane & 15, hi16 = (lane >> 4) << 4;
  const _Float16* arow  = h16 + (size_t)(base + mlane) * 256 + hi16;
  const _Float16* bbase = wsw + (size_t)(wv * 2) * 512 + lane * 16;
  v8f acc0 = {}, acc1 = {};
#pragma unroll
  for (int ks = 0; ks < 8; ++ks) {
    v16h a = ldfrag(arow + ks * 32);
    const _Float16* bp = bbase + (size_t)ks * (32 * 256);
    acc0 = wmma16(a, ldfrag(bp), acc0);
    acc1 = wmma16(a, ldfrag(bp + 512), acc1);
  }
  __syncthreads();  // all waves finished reading the block's h16 rows before in-place update
  int n0 = wv * 32 + mlane;
  float b0 = bself[n0], b1 = bself[n0 + 16];
  int moff = (lane >> 4) << 3;
  int p0 = apos(n0), p1 = apos(n0 + 16);
#pragma unroll
  for (int v = 0; v < 8; ++v) {
    int node = base + v + moff;
    float t0 = frelu(acc0[v] + b0 + agg[(size_t)node * 256 + n0]);
    float t1 = frelu(acc1[v] + b1 + agg[(size_t)node * 256 + n0 + 16]);
    h[(size_t)node * 256 + n0] = t0;
    h[(size_t)node * 256 + n0 + 16] = t1;
    h16[(size_t)node * 256 + p0] = (_Float16)t0;
    h16[(size_t)node * 256 + p1] = (_Float16)t1;
  }
}

// ---------------- degree (scatter count) ----------------
__global__ void k_degree(const int* __restrict__ edge_index, float* __restrict__ degree) {
  int e = blockIdx.x * 256 + threadIdx.x;
  if (e < N_EDGES) atomicAdd(&degree[edge_index[e]], 1.f);
}

// ---------------- valence violation damping ----------------
__global__ void k_violate(const float* __restrict__ degree, const int* __restrict__ pv,
                          float* __restrict__ h, _Float16* __restrict__ h16,
                          float* __restrict__ viol_out) {
  int idx = blockIdx.x * 256 + threadIdx.x;  // N_NODES * 256 threads
  int n = idx >> 8, c = idx & 255;
  float viol = frelu(degree[n] - (float)pv[n]);
  float s = 1.f / (1.f + viol);
  h[idx] = h[idx] * s;                                 // linear layout
  h16[idx] = (_Float16)((float)h16[idx] * s);          // swizzled layout (row-uniform scale)
  if (c == 0) viol_out[n] = viol;
}

// ---------------- bond MLP: [h[row]|h[col]|pv_r|pv_c](514->544) -> 128 -> 64 -> 4 ----------
__global__ void __launch_bounds__(256) k_bond(const _Float16* __restrict__ h16,
                                              const int* __restrict__ edge_index,
                                              const float* __restrict__ x,
                                              const int* __restrict__ pv,
                                              const _Float16* __restrict__ w1sw,
                                              const float* __restrict__ b1,
                                              const _Float16* __restrict__ w2sw,
                                              const float* __restrict__ b2,
                                              const float* __restrict__ w3,
                                              const float* __restrict__ b3,
                                              float* __restrict__ bond_out) {
  __shared__ __align__(32) _Float16 sA2[16 * 128];  // layer1 out, A-fragment order
  __shared__ _Float16 sA3[16 * 64];                 // layer2 out, linear (scalar tail)
  int tid = threadIdx.x, tile = blockIdx.x;
  int wv = tid >> 5, lane = tid & 31;
  int mlane = lane & 15, hi16 = (lane >> 4) << 4;
  int em = tile * 16 + mlane;
  int rnode = edge_index[em], cnode = edge_index[N_EDGES + em];
  const _Float16* arow  = h16 + (size_t)rnode * 256 + hi16;
  const _Float16* acol  = h16 + (size_t)cnode * 256 + hi16;
  const _Float16* bbase = w1sw + (size_t)wv * 512 + lane * 16;
  v8f acc = {};
#pragma unroll
  for (int ks = 0; ks < 16; ++ks) {
    v16h a = (ks < 8) ? ldfrag(arow + ks * 32) : ldfrag(acol + (ks - 8) * 32);
    acc = wmma16(a, ldfrag(bbase + (size_t)ks * (32 * 128)), acc);
  }
  {  // k-step 16: only k=512 (pv_row), k=513 (pv_col) nonzero -> register-built fragment
    v16h a = {};
    if (lane < 16) {
      a[0] = (_Float16)(float)pv[rnode];
      a[1] = (_Float16)(float)pv[cnode];
    }
    acc = wmma16(a, ldfrag(bbase + (size_t)16 * (32 * 128)), acc);
  }
  int col = wv * 16 + mlane;
  int moff = (lane >> 4) << 3;
  float bb = b1[col];
  int p = apos(col);
#pragma unroll
  for (int v = 0; v < 8; ++v)
    sA2[(v + moff) * 128 + p] = (_Float16)frelu(acc[v] + bb);
  __syncthreads();
  if (wv < 4) {  // wave-uniform: EXEC all-ones inside
    const _Float16* a2  = sA2 + mlane * 128 + hi16;
    const _Float16* b2p = w2sw + (size_t)wv * 512 + lane * 16;
    v8f acc2 = {};
#pragma unroll
    for (int ks = 0; ks < 4; ++ks)
      acc2 = wmma16(ldfrag(a2 + ks * 32), ldfrag(b2p + (size_t)ks * (32 * 64)), acc2);
    int c2 = wv * 16 + mlane;
    float bb2 = b2[c2];
#pragma unroll
    for (int v = 0; v < 8; ++v)
      sA3[(v + moff) * 64 + c2] = (_Float16)frelu(acc2[v] + bb2);
  }
  __syncthreads();
  if (tid < 64) {
    int le = tid >> 2, o = tid & 3;
    float s = b3[o];
#pragma unroll 8
    for (int k = 0; k < 64; ++k) s += (float)sA3[le * 64 + k] * w3[k * 4 + o];
    int r = edge_index[tile * 16 + le], c = edge_index[N_EDGES + tile * 16 + le];
    int atr = (int)x[r * 16]; atr = atr < 0 ? 0 : (atr > 10 ? 10 : atr);
    int atc = (int)x[c * 16]; atc = atc < 0 ? 0 : (atc > 10 ? 10 : atc);
    float halogen = (atr == 4 || atr == 5 || atc == 4 || atc == 5) ? 1.f : 0.f;
    int pr = pv[r], pc = pv[c];
    float lv1 = (pr <= 1 || pc <= 1) ? 1.f : 0.f;
    float lv2 = (pr <= 2 || pc <= 2) ? 1.f : 0.f;
    if (o >= 1) s += -100.f * halogen - 50.f * lv1;
    if (o == 2) s += -50.f * lv2;
    bond_out[(size_t)(tile * 16 + le) * 4 + o] = s;
  }
}

// ---------------- chem props MLP: h(256) -> 128 -> 32 ----------------
__global__ void __launch_bounds__(256) k_chem(const _Float16* __restrict__ h16,
                                              const _Float16* __restrict__ w1sw,
                                              const float* __restrict__ b1,
                                              const _Float16* __restrict__ w2sw,
                                              const float* __restrict__ b2,
                                              float* __restrict__ chem_out) {
  __shared__ __align__(32) _Float16 sA2[16 * 128];
  int tid = threadIdx.x, base = blockIdx.x * 16;
  int wv = tid >> 5, lane = tid & 31;
  int mlane = lane & 15, hi16 = (lane >> 4) << 4;
  const _Float16* arow  = h16 + (size_t)(base + mlane) * 256 + hi16;
  const _Float16* bbase = w1sw + (size_t)wv * 512 + lane * 16;
  v8f acc = {};
#pragma unroll
  for (int ks = 0; ks < 8; ++ks)
    acc = wmma16(ldfrag(arow + ks * 32), ldfrag(bbase + (size_t)ks * (32 * 128)), acc);
  int col = wv * 16 + mlane;
  int moff = (lane >> 4) << 3;
  float bb = b1[col];
  int p = apos(col);
#pragma unroll
  for (int v = 0; v < 8; ++v)
    sA2[(v + moff) * 128 + p] = (_Float16)frelu(acc[v] + bb);
  __syncthreads();
  if (wv < 2) {  // wave-uniform
    const _Float16* a2  = sA2 + mlane * 128 + hi16;
    const _Float16* b2p = w2sw + (size_t)wv * 512 + lane * 16;
    v8f acc2 = {};
#pragma unroll
    for (int ks = 0; ks < 4; ++ks)
      acc2 = wmma16(ldfrag(a2 + ks * 32), ldfrag(b2p + (size_t)ks * (32 * 32)), acc2);
    int c2 = wv * 16 + mlane;
    float bb2 = b2[c2];
#pragma unroll
    for (int v = 0; v < 8; ++v)
      chem_out[(size_t)(base + v + moff) * 32 + c2] = acc2[v] + bb2;
  }
}

extern "C" void kernel_launch(void* const* d_in, const int* in_sizes, int n_in,
                              void* d_out, int out_size, void* d_ws, size_t ws_size,
                              hipStream_t stream) {
  (void)in_sizes; (void)n_in; (void)out_size; (void)ws_size;
  const float* x         = (const float*)d_in[0];
  const float* edge_attr = (const float*)d_in[1];
  const float* atom_emb  = (const float*)d_in[2];
  const float* bond_tab  = (const float*)d_in[3];
  const float* vp_w1 = (const float*)d_in[4];
  const float* vp_b1 = (const float*)d_in[5];
  const float* vp_w2 = (const float*)d_in[6];
  const float* vp_b2 = (const float*)d_in[7];
  const float* gnn_wself = (const float*)d_in[8];
  const float* gnn_bself = (const float*)d_in[9];
  const float* gnn_wmsg  = (const float*)d_in[10];
  const float* gnn_bmsg  = (const float*)d_in[11];
  const float* bc_w1 = (const float*)d_in[12];
  const float* bc_b1 = (const float*)d_in[13];
  const float* bc_w2 = (const float*)d_in[14];
  const float* bc_b2 = (const float*)d_in[15];
  const float* bc_w3 = (const float*)d_in[16];
  const float* bc_b3 = (const float*)d_in[17];
  const float* cp_w1 = (const float*)d_in[18];
  const float* cp_b1 = (const float*)d_in[19];
  const float* cp_w2 = (const float*)d_in[20];
  const float* cp_b2 = (const float*)d_in[21];
  const int* edge_index = (const int*)d_in[22];

  float* out  = (float*)d_out;                 // outputs concat, return order
  float* h    = out;                           // [N,256]
  float* chem = out + (size_t)N_NODES * 256;   // [N,32]
  float* vlog = chem + (size_t)N_NODES * 32;   // [N,8]
  float* bond = vlog + (size_t)N_NODES * 8;    // [E,4]
  float* viol = bond + (size_t)N_EDGES * 4;    // [N]

  char* w = (char*)d_ws;
  size_t off = 0;
  _Float16* h16 = (_Float16*)(w + off);  off += (size_t)N_NODES * 256 * 2;   // 51.2 MB
  float* agg    = (float*)(w + off);     off += (size_t)N_NODES * 256 * 4;   // 102.4 MB
  int*   pv     = (int*)(w + off);       off += (size_t)N_NODES * 4;
  float* degree = (float*)(w + off);     off += (size_t)N_NODES * 4;
  _Float16* wmsg_sw  = (_Float16*)(w + off); off += (size_t)3 * 320 * 256 * 2;
  _Float16* wself_sw = (_Float16*)(w + off); off += (size_t)3 * 256 * 256 * 2;
  _Float16* bcw1_sw  = (_Float16*)(w + off); off += (size_t)544 * 128 * 2;
  _Float16* bcw2_sw  = (_Float16*)(w + off); off += (size_t)128 * 64 * 2;
  _Float16* cpw1_sw  = (_Float16*)(w + off); off += (size_t)256 * 128 * 2;
  _Float16* cpw2_sw  = (_Float16*)(w + off); off += (size_t)128 * 32 * 2;
  _Float16* be_sw    = (_Float16*)(w + off); off += (size_t)5 * 64 * 2;

  // ---- weight swizzle prep (tiny; L2-resident afterwards) ----
  for (int l = 0; l < 3; ++l) {
    k_prep_b<<<(320 * 256 + 255) / 256, 256, 0, stream>>>(
        gnn_wmsg + (size_t)l * 320 * 256, wmsg_sw + (size_t)l * 320 * 256, 320, 320, 256);
    k_prep_b<<<(256 * 256 + 255) / 256, 256, 0, stream>>>(
        gnn_wself + (size_t)l * 256 * 256, wself_sw + (size_t)l * 256 * 256, 256, 256, 256);
  }
  k_prep_b<<<(544 * 128 + 255) / 256, 256, 0, stream>>>(bc_w1, bcw1_sw, 514, 544, 128);
  k_prep_b<<<(128 * 64 + 255) / 256, 256, 0, stream>>>(bc_w2, bcw2_sw, 128, 128, 64);
  k_prep_b<<<(256 * 128 + 255) / 256, 256, 0, stream>>>(cp_w1, cpw1_sw, 256, 256, 128);
  k_prep_b<<<(128 * 32 + 255) / 256, 256, 0, stream>>>(cp_w2, cpw2_sw, 128, 128, 32);
  k_prep_be<<<2, 256, 0, stream>>>(bond_tab, be_sw);

  k_node_init<<<(N_NODES + 255) / 256, 256, 0, stream>>>(
      x, atom_emb, vp_w1, vp_b1, vp_w2, vp_b2, h, h16, pv, vlog);

  for (int l = 0; l < 3; ++l) {
    hipMemsetAsync(agg, 0, (size_t)N_NODES * 256 * 4, stream);
    k_msg<<<N_EDGES / 16, 256, 0, stream>>>(
        h16, edge_index, edge_attr, be_sw,
        wmsg_sw + (size_t)l * 320 * 256, gnn_bmsg + l * 256, agg);
    k_self<<<N_NODES / 16, 256, 0, stream>>>(
        agg, wself_sw + (size_t)l * 256 * 256, gnn_bself + l * 256, h, h16);
  }

  hipMemsetAsync(degree, 0, (size_t)N_NODES * 4, stream);
  k_degree<<<(N_EDGES + 255) / 256, 256, 0, stream>>>(edge_index, degree);
  k_violate<<<N_NODES, 256, 0, stream>>>(degree, pv, h, h16, viol);

  k_bond<<<N_EDGES / 16, 256, 0, stream>>>(
      h16, edge_index, x, pv, bcw1_sw, bc_b1, bcw2_sw, bc_b2, bc_w3, bc_b3, bond);
  k_chem<<<N_NODES / 16, 256, 0, stream>>>(h16, cpw1_sw, cp_b1, cpw2_sw, cp_b2, chem);
}
